// KernelUpdateHead_75325136437261
// MI455X (gfx1250) — matrix operand order repaired and checked by reading, not compiled, not run on hardware
//
#include <hip/hip_runtime.h>
#include <cstddef>

typedef __attribute__((ext_vector_type(8)))  float  v8f;
typedef __attribute__((ext_vector_type(16))) __bf16 v16bf;
typedef __attribute__((ext_vector_type(8)))  __bf16 v8bf;

#define WMMA_BF16(A, B, C) \
  __builtin_amdgcn_wmma_f32_16x16x32_bf16(false, (A), false, (B), (short)0, (C), false, false)

// ---------- helpers ----------
__device__ __forceinline__ __bf16 f2bf(float f) {
  return (__bf16)f;   // native v_cvt on gfx1250 (RNE)
}

__device__ __forceinline__ v16bf bzero16() {
  union { unsigned u[8]; v16bf v; } z;
#pragma unroll
  for (int i = 0; i < 8; ++i) z.u[i] = 0u;
  return z.v;
}

__device__ __forceinline__ float wsum32(float v) {
#pragma unroll
  for (int m = 16; m >= 1; m >>= 1) v += __shfl_xor(v, m, 32);
  return v;
}

// out[lane] = sum_k shfl(v,k) * W[lane*32+k]   (row-major W: (32,32))
__device__ __forceinline__ float mat32(float v, const float* W, int lane) {
  float acc = 0.f;
  const float* row = W + lane * 32;
#pragma unroll 8
  for (int k = 0; k < 32; ++k) acc += __shfl(v, k, 32) * row[k];
  return acc;
}

__device__ __forceinline__ float ln32(float v, const float* g, const float* b, int lane) {
  float mu = wsum32(v) * (1.f / 32.f);
  float d  = v - mu;
  float var = wsum32(d * d) * (1.f / 32.f);
  return d * rsqrtf(var + 1e-5f) * g[lane] + b[lane];
}

// ---------- tiny utility kernels ----------
__global__ void k_cvt_bf16(const float* __restrict__ s, __bf16* __restrict__ d, int n) {
  int i = blockIdx.x * 256 + threadIdx.x;
  if (i < n) d[i] = f2bf(s[i]);
}

// features (B,C,HW) f32 -> featT (B,HW,C) bf16
__global__ void k_transpose_feat(const float* __restrict__ feats, __bf16* __restrict__ featT) {
  int t = blockIdx.x * 256 + threadIdx.x;          // b*25600 + p, total 204800
  if (t >= 8 * 25600) return;
  int b = t / 25600, p = t % 25600;
  const float* src = feats + (size_t)b * 32 * 25600 + p;
  __bf16* dst = featT + (size_t)t * 32;
#pragma unroll
  for (int c = 0; c < 32; ++c) dst[c] = f2bf(src[(size_t)c * 25600]);
}

// ---------- stage 1: fk = einsum('bnhw,bchw->bnc'), K split 16 ways ----------
// featC: features pre-converted to bf16, same (B,C,HW) layout
__global__ void k_stage1(const __bf16* __restrict__ featC, const float* __restrict__ imaps,
                         float* __restrict__ fkpart) {
  int gid = blockIdx.x;                 // B*8*16 blocks, 32 threads (one wave)
  int ks = gid & 15; gid >>= 4;
  int mt = gid & 7;  int b = gid >> 3;
  int lane = threadIdx.x;
  int L = lane & 15, g = lane >> 4;
  int n = mt * 16 + L;
  const float*  arow = imaps + ((size_t)(b * 128 + n)) * 25600;
  const __bf16* c0p  = featC + ((size_t)(b * 32 + L)) * 25600;
  const __bf16* c1p  = featC + ((size_t)(b * 32 + 16 + L)) * 25600;
  v8f acc0 = {}; v8f acc1 = {};
  int kbase = ks * 50 * 32;
  for (int kt = 0; kt < 50; ++kt) {
    int k0 = kbase + kt * 32;
    __builtin_prefetch(arow + k0 + 1024, 0, 3);   // stream imaps ahead (global_prefetch_b8)
    v16bf a = bzero16();
#pragma unroll
    for (int j = 0; j < 8; ++j) {
      a[j]     = f2bf(arow[k0 + 16 * g + j]);
      a[8 + j] = f2bf(arow[k0 + 16 * g + 8 + j]);
    }
    v16bf b0 = *(const v16bf*)(c0p + k0 + 16 * g);
    v16bf b1 = *(const v16bf*)(c1p + k0 + 16 * g);
    acc0 = WMMA_BF16(a, b0, acc0);
    acc1 = WMMA_BF16(a, b1, acc1);
  }
  float* dst = fkpart + (size_t)ks * 32768;
#pragma unroll
  for (int r = 0; r < 8; ++r) {
    int row = b * 128 + mt * 16 + r + 8 * g;
    dst[row * 32 + L]      = acc0[r];
    dst[row * 32 + 16 + L] = acc1[r];
  }
}

__global__ void k_reduce_fk(const float* __restrict__ fkpart, float* __restrict__ fk) {
  int t = blockIdx.x * 256 + threadIdx.x;
  if (t >= 32768) return;
  float s = 0.f;
#pragma unroll
  for (int ks = 0; ks < 16; ++ks) s += fkpart[(size_t)ks * 32768 + t];
  fk[t] = s;
}

// ---------- stage 2: small dense chain ----------
__global__ void k_params(const float* __restrict__ fk, const float* __restrict__ dl_w,
                         const float* __restrict__ dl_b, float* __restrict__ params) {
  int t = blockIdx.x * 256 + threadIdx.x;          // 1024*64
  if (t >= 65536) return;
  int j = t & 63, bn = t >> 6;
  const float* fr = fk + bn * 32;
  const float* wr = dl_w + j * 32;
  float acc = dl_b[j];
#pragma unroll 8
  for (int c = 0; c < 32; ++c) acc += fr[c] * wr[c];
  params[t] = acc;
}

__global__ void k_ifeat(const float* __restrict__ kernels, const float* __restrict__ il_w,
                        const float* __restrict__ il_b, float* __restrict__ ifeat) {
  int t = blockIdx.x * 256 + threadIdx.x;          // 1024*9*64 = 589824
  if (t >= 589824) return;
  int j = t % 64; int kk = (t / 64) % 9; int bn = t / 576;
  const float* kr = kernels + (size_t)bn * 288 + kk;   // (bn, c, kk): stride 9 in c
  const float* wr = il_w + j * 32;
  float acc = il_b[j];
#pragma unroll 8
  for (int c = 0; c < 32; ++c) acc += kr[c * 9] * wr[c];
  ifeat[t] = acc;
}

__global__ void k_gates(const float* __restrict__ params, const float* __restrict__ ifeat,
                        const float* __restrict__ ig_w, const float* __restrict__ ig_b,
                        const float* __restrict__ ug_w, const float* __restrict__ ug_b,
                        const float* __restrict__ inin_g, const float* __restrict__ inin_b,
                        const float* __restrict__ nin_g,  const float* __restrict__ nin_b,
                        const float* __restrict__ nout_g, const float* __restrict__ nout_b,
                        const float* __restrict__ inout_g,const float* __restrict__ inout_b,
                        const float* __restrict__ fc_w,  const float* __restrict__ fc_b,
                        const float* __restrict__ fcn_g, const float* __restrict__ fcn_b,
                        __bf16* __restrict__ xbf) {
  int row = blockIdx.x * 8 + (threadIdx.x >> 5);   // 9216 rows
  int lane = threadIdx.x & 31;
  if (row >= 9216) return;
  int bn = row / 9, kk = row % 9;
  float pin  = params[bn * 64 + lane];
  float pout = params[bn * 64 + 32 + lane];
  float iin  = ifeat[(size_t)row * 64 + lane];
  float iout = ifeat[(size_t)row * 64 + 32 + lane];
  float gate = iin * pin;
  float ig = 1.f / (1.f + __expf(-ln32(mat32(gate, ig_w, lane) + ig_b[lane], inin_g, inin_b, lane)));
  float ug = 1.f / (1.f + __expf(-ln32(mat32(gate, ug_w, lane) + ug_b[lane], nin_g,  nin_b,  lane)));
  float po = fmaxf(ln32(pout, nout_g,  nout_b,  lane), 0.f);
  float io = fmaxf(ln32(iout, inout_g, inout_b, lane), 0.f);
  float kb = ug * po + ig * io;
  float xv = fmaxf(ln32(mat32(kb, fc_w, lane) + fc_b[lane], fcn_g, fcn_b, lane), 0.f);
  xbf[(size_t)bn * 288 + kk * 32 + lane] = f2bf(xv);
}

// ---------- generic bf16 GEMM: out = X(M,K) @ Wt(N,K)^T + bias, out f32 ----------
__global__ void k_gemm_xwt(const __bf16* __restrict__ X, const __bf16* __restrict__ Wt,
                           const float* __restrict__ bias, float* __restrict__ out,
                           int M, int Nn, int Kd) {
  int wave = threadIdx.x >> 5, lane = threadIdx.x & 31;
  int mt = blockIdx.x * 2 + (wave & 1);
  int nt = blockIdx.y * 4 + (wave >> 1);
  int m0 = mt * 16, n0 = nt * 16;
  if (m0 >= M || n0 >= Nn) return;
  int L = lane & 15, g = lane >> 4;
  v8f acc = {};
  const __bf16* arow = X + (size_t)(m0 + L) * Kd;
  const __bf16* brow = Wt + (size_t)(n0 + L) * Kd;
  for (int k0 = 0; k0 < Kd; k0 += 32) {
    union { v16bf v; v8bf h[2]; } a;
    a.h[0] = *(const v8bf*)(arow + k0 + 16 * g);
    a.h[1] = *(const v8bf*)(arow + k0 + 16 * g + 8);
    v16bf b = *(const v16bf*)(brow + k0 + 16 * g);
    acc = WMMA_BF16(a.v, b, acc);
  }
#pragma unroll
  for (int r = 0; r < 8; ++r) {
    int m = m0 + r + 8 * g, n = n0 + L;
    out[(size_t)m * Nn + n] = acc[r] + bias[n];
  }
}

// ---------- attention core: scores + softmax + AV per (b, head) ----------
__global__ void k_attn(const float* __restrict__ qkv, __bf16* __restrict__ obf) {
  __shared__ float S[128][128];                       // exactly 64 KB
  int h = blockIdx.x & 7, b = blockIdx.x >> 3;
  int t = threadIdx.x;
  const float* base = qkv + (size_t)b * 128 * 864 + h * 36;
  for (int idx = t; idx < 16384; idx += 256) {
    int i = idx >> 7, j = idx & 127;
    const float* qi = base + (size_t)i * 864;
    const float* kj = base + (size_t)j * 864 + 288;
    float acc = 0.f;
#pragma unroll 9
    for (int d = 0; d < 36; ++d) acc += qi[d] * kj[d];
    S[i][j] = acc * (1.0f / 6.0f);                    // 1/sqrt(36)
  }
  __syncthreads();
  if (t < 128) {
    int i = t;
    float mx = -1e30f;
    for (int jo = 0; jo < 128; ++jo) { int j = (jo + i) & 127; mx = fmaxf(mx, S[i][j]); }
    float sum = 0.f;
    for (int jo = 0; jo < 128; ++jo) { int j = (jo + i) & 127; float e = __expf(S[i][j] - mx); S[i][j] = e; sum += e; }
    float inv = 1.f / sum;
    for (int jo = 0; jo < 128; ++jo) { int j = (jo + i) & 127; S[i][j] *= inv; }
  }
  __syncthreads();
  for (int idx = t; idx < 4608; idx += 256) {         // 128 * 36
    int i = idx / 36, d = idx % 36;
    float acc = 0.f;
    for (int jo = 0; jo < 128; ++jo) {
      int j = (jo + i) & 127;
      acc += S[i][j] * base[(size_t)j * 864 + 576 + d];
    }
    obf[(size_t)(b * 128 + i) * 288 + h * 36 + d] = f2bf(acc);
  }
}

// ---------- LN(an) + emit k_new ----------
__global__ void k_ln_an(const float* __restrict__ x2, const float* __restrict__ an_g,
                        const float* __restrict__ an_b, float* __restrict__ kbar2,
                        float* __restrict__ kout) {
  int row = blockIdx.x * 8 + (threadIdx.x >> 5);      // 1024 rows
  int lane = threadIdx.x & 31;
  if (row >= 1024) return;
  const float* xr = x2 + (size_t)row * 288;
  float v[9]; float s = 0.f;
#pragma unroll
  for (int i = 0; i < 9; ++i) { v[i] = xr[lane + 32 * i]; s += v[i]; }
  float mu = wsum32(s) * (1.f / 288.f);
  float s2 = 0.f;
#pragma unroll
  for (int i = 0; i < 9; ++i) { float d = v[i] - mu; s2 += d * d; }
  float rs = rsqrtf(wsum32(s2) * (1.f / 288.f) + 1e-5f);
#pragma unroll
  for (int i = 0; i < 9; ++i) {
    int e = lane + 32 * i;                            // c = lane, kk = i
    float o = (v[i] - mu) * rs * an_g[e] + an_b[e];
    kbar2[(size_t)row * 288 + e] = o;
    kout[((size_t)row * 32 + lane) * 9 + i] = o;      // k_new (B,N,C,3,3)
  }
}

// ---------- reg + fm fused -> dynamic conv weights (bf16) ----------
__global__ void k_regfm(const float* __restrict__ kbar2,
                        const float* __restrict__ reg_w, const float* __restrict__ reg_b,
                        const float* __restrict__ regn_g, const float* __restrict__ regn_b,
                        const float* __restrict__ fm_w, const float* __restrict__ fm_b,
                        __bf16* __restrict__ mbuf) {
  int row = blockIdx.x * 8 + (threadIdx.x >> 5);      // 9216 rows
  int lane = threadIdx.x & 31;
  if (row >= 9216) return;
  float kb = kbar2[(size_t)row * 32 + lane];
  float r = fmaxf(ln32(mat32(kb, reg_w, lane) + reg_b[lane], regn_g, regn_b, lane), 0.f);
  float m = mat32(r, fm_w, lane) + fm_b[lane];
  mbuf[(size_t)row * 32 + lane] = f2bf(m);
}

// ---------- fused dynamic 3x3 grouped conv (WMMA) + softmax over N ----------
__global__ void k_conv_softmax(const __bf16* __restrict__ featT, const __bf16* __restrict__ mbuf,
                               float* __restrict__ out) {
  __shared__ float sm[128][16];
  __shared__ float pmax[16][16];
  __shared__ float psum[16][16];
  __shared__ float cred[2][16];
  int blk = blockIdx.x;                               // 8 * 1600
  int tile = blk % 1600, b = blk / 1600;
  int wave = threadIdx.x >> 5, lane = threadIdx.x & 31;
  int L = lane & 15, g = lane >> 4;
  int p0 = tile * 16;
  int p = p0 + L;
  int h = p / 160, w = p % 160;
  v8f acc = {};
  const __bf16* mrow = mbuf + (size_t)(b * 128 + wave * 16 + L) * 288;
#pragma unroll
  for (int kk = 0; kk < 9; ++kk) {
    int dh = kk / 3 - 1, dw = kk % 3 - 1;
    union { v16bf v; v8bf hlf[2]; } a;
    const __bf16* ap = mrow + kk * 32;
    a.hlf[0] = *(const v8bf*)(ap + 16 * g);
    a.hlf[1] = *(const v8bf*)(ap + 16 * g + 8);
    int hh = h + dh, ww = w + dw;
    v16bf bf = bzero16();
    if (hh >= 0 && hh < 160 && ww >= 0 && ww < 160) {
      bf = *(const v16bf*)(featT + ((size_t)(b * 25600 + hh * 160 + ww)) * 32 + 16 * g);
    }
    acc = WMMA_BF16(a.v, bf, acc);
  }
#pragma unroll
  for (int r = 0; r < 8; ++r) sm[wave * 16 + r + 8 * g][L] = acc[r];
  __syncthreads();

  int col = threadIdx.x & 15, seg = threadIdx.x >> 4;
  float mx = -1e30f;
#pragma unroll
  for (int i = 0; i < 8; ++i) mx = fmaxf(mx, sm[seg * 8 + i][col]);
  pmax[seg][col] = mx;
  __syncthreads();
  if (threadIdx.x < 16) {
    float m2 = pmax[0][threadIdx.x];
#pragma unroll
    for (int s = 1; s < 16; ++s) m2 = fmaxf(m2, pmax[s][threadIdx.x]);
    cred[0][threadIdx.x] = m2;
  }
  __syncthreads();
  float cm = cred[0][col];
  float ssum = 0.f;
#pragma unroll
  for (int i = 0; i < 8; ++i) {
    float e = __expf(sm[seg * 8 + i][col] - cm);
    sm[seg * 8 + i][col] = e;
    ssum += e;
  }
  psum[seg][col] = ssum;
  __syncthreads();
  if (threadIdx.x < 16) {
    float s2 = 0.f;
#pragma unroll
    for (int s = 0; s < 16; ++s) s2 += psum[s][threadIdx.x];
    cred[1][threadIdx.x] = s2;
  }
  __syncthreads();
  float inv = 1.f / cred[1][col];
#pragma unroll
  for (int i = 0; i < 8; ++i) {
    int row = seg * 8 + i;
    out[(size_t)(b * 128 + row) * 25600 + p0 + col] = sm[row][col] * inv;
  }
}

// ---------- host launcher ----------
extern "C" void kernel_launch(void* const* d_in, const int* in_sizes, int n_in,
                              void* d_out, int out_size, void* d_ws, size_t ws_size,
                              hipStream_t stream) {
  (void)in_sizes; (void)n_in; (void)out_size; (void)ws_size;
  const float* features   = (const float*)d_in[0];
  const float* kernels    = (const float*)d_in[1];
  const float* imaps      = (const float*)d_in[2];
  const float* dl_w       = (const float*)d_in[3];
  const float* dl_b       = (const float*)d_in[4];
  const float* il_w       = (const float*)d_in[5];
  const float* il_b       = (const float*)d_in[6];
  const float* ig_w       = (const float*)d_in[7];
  const float* ig_b       = (const float*)d_in[8];
  const float* ug_w       = (const float*)d_in[9];
  const float* ug_b       = (const float*)d_in[10];
  const float* inin_g     = (const float*)d_in[11];
  const float* inin_b     = (const float*)d_in[12];
  const float* inout_g    = (const float*)d_in[13];
  const float* inout_b    = (const float*)d_in[14];
  const float* nin_g      = (const float*)d_in[15];
  const float* nin_b      = (const float*)d_in[16];
  const float* nout_g     = (const float*)d_in[17];
  const float* nout_b     = (const float*)d_in[18];
  const float* fc_w       = (const float*)d_in[19];
  const float* fc_b       = (const float*)d_in[20];
  const float* fcn_g      = (const float*)d_in[21];
  const float* fcn_b      = (const float*)d_in[22];
  const float* attn_in_w  = (const float*)d_in[23];
  const float* attn_in_b  = (const float*)d_in[24];
  const float* attn_out_w = (const float*)d_in[25];
  const float* attn_out_b = (const float*)d_in[26];
  const float* an_g       = (const float*)d_in[27];
  const float* an_b       = (const float*)d_in[28];
  const float* reg_w      = (const float*)d_in[29];
  const float* reg_b      = (const float*)d_in[30];
  const float* regn_g     = (const float*)d_in[31];
  const float* regn_b     = (const float*)d_in[32];
  const float* fm_w       = (const float*)d_in[33];
  const float* fm_b       = (const float*)d_in[34];

  float* out_maps = (float*)d_out;                      // (B,N,H,W) 26,214,400
  float* out_knew = (float*)d_out + 26214400;           // (B,N,C,3,3) 294,912

  char* ws = (char*)d_ws;
  size_t off = 0;
  auto alloc = [&](size_t bytes) -> void* {
    void* p = ws + off;
    off += (bytes + 255) & ~(size_t)255;
    return p;
  };
  float*  fkpart = (float*)alloc(16 * 32768 * sizeof(float));
  float*  fk     = (float*)alloc(32768 * sizeof(float));
  float*  params = (float*)alloc(65536 * sizeof(float));
  float*  ifeat  = (float*)alloc(589824 * sizeof(float));
  __bf16* xbf    = (__bf16*)alloc(294912 * sizeof(__bf16));
  float*  qkv    = (float*)alloc(884736 * sizeof(float));
  __bf16* obf    = (__bf16*)alloc(294912 * sizeof(__bf16));
  float*  x2     = (float*)alloc(294912 * sizeof(float));
  float*  kbar2  = (float*)alloc(294912 * sizeof(float));
  __bf16* mbuf   = (__bf16*)alloc(294912 * sizeof(__bf16));
  __bf16* featT  = (__bf16*)alloc((size_t)8 * 25600 * 32 * sizeof(__bf16));  // (B,HW,C)
  __bf16* featC  = (__bf16*)alloc((size_t)8 * 32 * 25600 * sizeof(__bf16));  // (B,C,HW)
  __bf16* wqkv   = (__bf16*)alloc((size_t)864 * 288 * sizeof(__bf16));
  __bf16* wout   = (__bf16*)alloc((size_t)288 * 288 * sizeof(__bf16));

  // weight conversions + feature layouts (one pass each)
  k_cvt_bf16<<<(248832 + 255) / 256, 256, 0, stream>>>(attn_in_w, wqkv, 248832);
  k_cvt_bf16<<<(82944 + 255) / 256, 256, 0, stream>>>(attn_out_w, wout, 82944);
  k_cvt_bf16<<<(6553600 + 255) / 256, 256, 0, stream>>>(features, featC, 6553600);
  k_transpose_feat<<<800, 256, 0, stream>>>(features, featT);

  // stage 1: bnc einsum via WMMA, K split 16 ways, deterministic reduce
  k_stage1<<<8 * 8 * 16, 32, 0, stream>>>(featC, imaps, fkpart);
  k_reduce_fk<<<128, 256, 0, stream>>>(fkpart, fk);

  // stage 2: small dense chain
  k_params<<<256, 256, 0, stream>>>(fk, dl_w, dl_b, params);
  k_ifeat<<<2304, 256, 0, stream>>>(kernels, il_w, il_b, ifeat);
  k_gates<<<1152, 256, 0, stream>>>(params, ifeat, ig_w, ig_b, ug_w, ug_b,
                                    inin_g, inin_b, nin_g, nin_b, nout_g, nout_b,
                                    inout_g, inout_b, fc_w, fc_b, fcn_g, fcn_b, xbf);

  // attention: QKV (WMMA), core (LDS softmax), out-proj (WMMA), LN + k_new
  k_gemm_xwt<<<dim3(32, 14), 256, 0, stream>>>(xbf, wqkv, attn_in_b, qkv, 1024, 864, 288);
  k_attn<<<64, 256, 0, stream>>>(qkv, obf);
  k_gemm_xwt<<<dim3(32, 5), 256, 0, stream>>>(obf, wout, attn_out_b, x2, 1024, 288, 288);
  k_ln_an<<<128, 256, 0, stream>>>(x2, an_g, an_b, kbar2, out_knew);

  // dynamic weights, then fused conv + softmax (WMMA)
  k_regfm<<<1152, 256, 0, stream>>>(kbar2, reg_w, reg_b, regn_g, regn_b, fm_w, fm_b, mbuf);
  k_conv_softmax<<<8 * 1600, 256, 0, stream>>>(featT, mbuf, out_maps);
}
